// Model_RK_67379446939985
// MI455X (gfx1250) — matrix operand order repaired and checked by reading, not compiled, not run on hardware
//
#include <hip/hip_runtime.h>
#include <hip/hip_bf16.h>
#include <cstdint>
#include <cstddef>

#define BATCH   2048
#define NSTEP   128
#define DDIM    100
#define XDIM    100
#define HDIM    256
#define K0      128     // padded MLP input dim (101 -> 128)
#define NOUT    112     // padded MLP output dim (100 -> 112 = 7*16)

#define DT      (1.0f/128.0f)
#define SQDT    0.088388347648318447f   // sqrt(1/128)
#define SQRT2   1.4142135623730951f

typedef __attribute__((ext_vector_type(16))) __bf16 v16bf;
typedef __attribute__((ext_vector_type(8)))  __bf16 v8bf;
typedef __attribute__((ext_vector_type(8)))  float  v8f;
typedef __attribute__((ext_vector_type(4)))  unsigned int v4u;
typedef __attribute__((ext_vector_type(8)))  int v8i;
typedef __attribute__((ext_vector_type(4)))  int v4i;

union Frag32B { v16bf v; v8bf h[2]; };

// ---------------------------------------------------------------------------
// prep_x: prefix-scan of noise per (b,d); writes bf16 activation rows for both
// phi evals of every step, and the final x output (x_T = x0 + sqrt2*sum w).
// Row index in U: r = (step*BATCH + b)*2 + phi, each row K0 bf16 wide.
// ---------------------------------------------------------------------------
__global__ void prep_x(const float* __restrict__ wn, const int* __restrict__ signs,
                       const float* __restrict__ x0, __bf16* __restrict__ U,
                       float* __restrict__ xout)
{
  int tid = blockIdx.x * blockDim.x + threadIdx.x;
  if (tid >= BATCH * DDIM) return;
  int b = tid / DDIM, d = tid % DDIM;
  const float* wp = wn + (size_t)(b * DDIM + d) * NSTEP;
  float x = x0[d];
  for (int i = 0; i < NSTEP; ++i) {
    float w   = wp[i] * SQDT;
    float sdt = (float)(2 * signs[i] - 1) * SQDT;
    size_t r0 = ((size_t)(i * BATCH + b)) * 2u;
    U[r0 * K0 + d]       = (__bf16)x;                       // phi0: x_i
    U[(r0 + 1) * K0 + d] = (__bf16)(x + SQRT2 * (w - sdt)); // phi1: x_i + k1
    x += SQRT2 * w;
  }
  xout[b * XDIM + d] = x;
}

// ---------------------------------------------------------------------------
// prep_cols: time column (col 100) + zero padding (cols 101..127) per row.
// ---------------------------------------------------------------------------
__global__ void prep_cols(__bf16* __restrict__ U)
{
  int r = blockIdx.x * blockDim.x + threadIdx.x;
  if (r >= NSTEP * BATCH * 2) return;
  int i   = r >> 12;          // / (BATCH*2)
  int phi = r & 1;
  size_t base = (size_t)r * K0;
  U[base + 100] = (__bf16)(DT * (float)(i + phi));
  #pragma unroll
  for (int c = 101; c < K0; ++c) U[base + c] = (__bf16)0.0f;
}

// ---------------------------------------------------------------------------
// prep_w: repack fp32 weights into bf16 WMMA-B tiles (32x16, K-major lanes).
// Tile element (l, h) holds W[kt*32 + 16*(l>=16) + h][nt*16 + (l&15)].
// Layer bases (bf16 elems): L1=0 (4x16 tiles), L2=32768, L3=98304, L4=163840.
// ---------------------------------------------------------------------------
__global__ void prep_w(const float* __restrict__ Wi,  const float* __restrict__ Wh1,
                       const float* __restrict__ Wh2, const float* __restrict__ Wo,
                       __bf16* __restrict__ WPK)
{
  int gid = blockIdx.x * blockDim.x + threadIdx.x;
  if (gid >= 192512) return;
  const float* src; int nNt, kvalid, nvalid, srcN; int idx = gid;
  if (idx < 32768)       {                src = Wi;  nNt = 16; srcN = 256; kvalid = 101; nvalid = 256; }
  else if (idx < 98304)  { idx -= 32768;  src = Wh1; nNt = 16; srcN = 256; kvalid = 256; nvalid = 256; }
  else if (idx < 163840) { idx -= 98304;  src = Wh2; nNt = 16; srcN = 256; kvalid = 256; nvalid = 256; }
  else                   { idx -= 163840; src = Wo;  nNt = 7;  srcN = 100; kvalid = 256; nvalid = 100; }
  int tl = idx >> 9, e = idx & 511;
  int l = e >> 4, h = e & 15;
  int kt = tl / nNt, nt = tl % nNt;
  int k = kt * 32 + ((l >= 16) ? 16 : 0) + h;
  int n = nt * 16 + (l & 15);
  float v = (k < kvalid && n < nvalid) ? src[k * srcN + n] : 0.0f;
  WPK[gid] = (__bf16)v;
}

// ---------------------------------------------------------------------------
// TDM staging: one Tensor-DMA descriptor per wave pulls its 16x128 bf16 tile
// (global row stride = 256 elems, i.e. skipping the interleaved phi row)
// straight into the wave's LDS activation region.
// This toolchain exposes the 6-arg tensor_load_to_lds builtin:
//   (uint32x4 g0, int32x8 g1, int32x4 g2, int32x4 g3, int32x8 g4, i32 cpol)
// ---------------------------------------------------------------------------
__device__ __forceinline__ void stage_tile_to_lds(__bf16* actw, const __bf16* grow)
{
#if __has_builtin(__builtin_amdgcn_tensor_load_to_lds)
  unsigned lds_off = (unsigned)(size_t)(const void*)actw;
  unsigned long long ga = (unsigned long long)(const void*)grow;
  v4u g0;
  g0[0] = 1u;                                                 // count=1, user D#
  g0[1] = lds_off;                                            // lds_addr (bytes)
  g0[2] = (unsigned)(ga & 0xFFFFFFFFu);                       // global_addr[31:0]
  g0[3] = (unsigned)((ga >> 32) & 0x1FFFFFFu) | (2u << 30);   // addr[56:32] | type=2
  v8i g1;
  g1[0] = (int)(1u << 16);        // workgroup_mask=0, data_size=1 (2-byte elems)
  g1[1] = (int)(256u << 16);      // tensor_dim0 = 256 elems (low 16 bits)
  g1[2] = (int)(16u << 16);       // tensor_dim0 hi=0 | tensor_dim1 = 16
  g1[3] = (int)(128u << 16);      // tensor_dim1 hi=0 | tile_dim0 = 128
  g1[4] = 16;                     // tile_dim1 = 16, tile_dim2 = 0
  g1[5] = 256;                    // tensor_dim0_stride = 256 elems (2 U rows)
  g1[6] = 0;
  g1[7] = 0;
  v4i gz4 = {0, 0, 0, 0};
  v8i gz8 = {0, 0, 0, 0, 0, 0, 0, 0};
  __builtin_amdgcn_tensor_load_to_lds(g0, g1, gz4, gz4, gz8, 0);
  __builtin_amdgcn_s_wait_tensorcnt(0);
  asm volatile("" ::: "memory");
#else
  // fallback: async gather of 16B chunks into LDS
  const int lane = (int)(threadIdx.x & 31);
  const int mrow = lane & 15;
  unsigned long long g0 = (unsigned long long)(const void*)(grow + (size_t)mrow * 256);
  unsigned l0 = (unsigned)(size_t)(const void*)(actw + mrow * 256);
  #pragma unroll
  for (int j = 0; j < 8; ++j) {
    int chunk = (lane >> 4) + 2 * j;
    unsigned long long ga = g0 + (unsigned long long)(chunk * 16);
    unsigned la = l0 + (unsigned)(chunk * 16);
    asm volatile("global_load_async_to_lds_b128 %0, %1, off" :: "v"(la), "v"(ga) : "memory");
  }
  asm volatile("s_wait_asynccnt 0" ::: "memory");
#endif
}

// ---------------------------------------------------------------------------
// Core GEMM tile loop: A fragments from this wave's LDS activation rows
// (bf16 A layout: lane = row l&15; K chunks split by lane half), B fragments
// streamed from the packed weight buffer (L2-resident).
// ---------------------------------------------------------------------------
template<int NKT, int NNT>
__device__ __forceinline__ void gemm_tiles(const __bf16* actw, int lane,
                                           const __bf16* __restrict__ wpk,
                                           v8f* acc)
{
  const int sub  = (lane & 16) ? 8 : 0;
  const int arow = lane & 15;
  for (int kt = 0; kt < NKT; ++kt) {
    Frag32B a;
    const __bf16* rp = actw + arow * 256 + kt * 32;
    a.h[0] = *(const v8bf*)(const void*)(rp + 2 * sub);
    a.h[1] = *(const v8bf*)(const void*)(rp + 2 * sub + 8);
    #pragma unroll
    for (int nt = 0; nt < NNT; ++nt) {
      Frag32B bm;
      const __bf16* wp = wpk + (size_t)(kt * NNT + nt) * 512 + lane * 16;
      bm.h[0] = *(const v8bf*)(const void*)(wp);
      bm.h[1] = *(const v8bf*)(const void*)(wp + 8);
      acc[nt] = __builtin_amdgcn_wmma_f32_16x16x32_bf16(false, a.v, false, bm.v,
                                                        (short)0, acc[nt], false, false);
    }
  }
}

// Hidden layer: zero-init acc, GEMM, bias+ReLU in epilogue, bf16 back to LDS.
template<int NKT>
__device__ __forceinline__ void hidden_layer(__bf16* actw, int lane,
                                             const __bf16* __restrict__ wpk,
                                             const float* __restrict__ bias)
{
  const int col   = lane & 15;
  const int rowhi = (lane >> 4) << 3;
  v8f acc[16] = {};
  gemm_tiles<NKT, 16>(actw, lane, wpk, acc);
  #pragma unroll
  for (int nt = 0; nt < 16; ++nt) {
    float bv = bias[nt * 16 + col];
    #pragma unroll
    for (int g = 0; g < 8; ++g)
      actw[(g + rowhi) * 256 + nt * 16 + col] = (__bf16)fmaxf(acc[nt][g] + bv, 0.0f);
  }
}

// ---------------------------------------------------------------------------
// Main fused MLP kernel. Grid: 128 steps * 2 phis * 16 batch tiles = 4096 WGs.
// 8 waves/WG; each wave owns 16 batch rows end-to-end (no cross-wave deps).
// ---------------------------------------------------------------------------
__launch_bounds__(256, 1)
__global__ void bsde_main(const __bf16* __restrict__ U,
                          const __bf16* __restrict__ WPK,
                          const float* __restrict__ wn,
                          const int* __restrict__ signs,
                          const float* __restrict__ bi,
                          const float* __restrict__ bh1,
                          const float* __restrict__ bh2,
                          const float* __restrict__ bo,
                          float* __restrict__ part)
{
  __shared__ __bf16 act[8][16 * 256];               // 64 KB: 16 rows x 256 bf16 per wave
  const int wave = threadIdx.x >> 5, lane = threadIdx.x & 31;
  const int bid   = blockIdx.x;
  const int step  = bid >> 5, rem = bid & 31;
  const int phi   = rem >> 4, btile = rem & 15;
  const int rbase = btile * 128 + wave * 16;        // batch-row base of this wave
  __bf16* actw = &act[wave][0];

  // ---- TDM-stage the wave's 16x128 bf16 input tile into LDS ----
  stage_tile_to_lds(actw, U + ((size_t)((step * BATCH + rbase) * 2 + phi)) * K0);

  const int col   = lane & 15;
  const int rowhi = (lane >> 4) << 3;

  hidden_layer<4>(actw, lane, WPK,          bi);    // [16x128]@[128x256]
  hidden_layer<8>(actw, lane, WPK + 32768,  bh1);   // [16x256]@[256x256]
  hidden_layer<8>(actw, lane, WPK + 98304,  bh2);   // [16x256]@[256x256]

  // ---- layer 4: [16x256] @ [256x112]; bo folded into the reduction ----
  v8f z[7] = {};
  gemm_tiles<8, 7>(actw, lane, WPK + 163840, z);

  // ---- per-row reduction: c = [phi0] -dt*sum(z^2) + 0.5*z.(w-sdt)
  //                             [phi1]                0.5*z.(w+sdt)
  float s[8];
  #pragma unroll
  for (int g = 0; g < 8; ++g) s[g] = 0.0f;
  const float sdt  = (float)(2 * signs[step] - 1) * SQDT;
  const float wsgn = phi ? 1.0f : -1.0f;
  #pragma unroll
  for (int nt = 0; nt < 7; ++nt) {
    int d = nt * 16 + col;
    if (d < DDIM) {
      float bv = bo[d];
      #pragma unroll
      for (int g = 0; g < 8; ++g) {
        int b = rbase + g + rowhi;
        float zv = z[nt][g] + bv;
        float w  = wn[((size_t)(b * DDIM + d)) * NSTEP + step] * SQDT;
        float wv = w + wsgn * sdt;
        float c  = 0.5f * zv * wv;
        if (phi == 0) c -= DT * zv * zv;
        s[g] += c;
      }
    }
  }
  // reduce over the 16 lanes of each half (xor masks < 16 stay within half)
  #pragma unroll
  for (int g = 0; g < 8; ++g) {
    s[g] += __shfl_xor(s[g], 1, 32);
    s[g] += __shfl_xor(s[g], 2, 32);
    s[g] += __shfl_xor(s[g], 4, 32);
    s[g] += __shfl_xor(s[g], 8, 32);
  }
  if (col == 0) {
    #pragma unroll
    for (int g = 0; g < 8; ++g)
      part[(size_t)(step * 2 + phi) * BATCH + (rbase + g + rowhi)] = s[g];
  }
}

// ---------------------------------------------------------------------------
// y_reduce: y[b] = y0 + sum over 256 (step,phi) partials.
// ---------------------------------------------------------------------------
__global__ void y_reduce(const float* __restrict__ part, const float* __restrict__ y0,
                         float* __restrict__ yout)
{
  int b = blockIdx.x * blockDim.x + threadIdx.x;
  if (b >= BATCH) return;
  float acc = y0[0];
  for (int s = 0; s < NSTEP * 2; ++s) acc += part[(size_t)s * BATCH + b];
  yout[b] = acc;
}

// ---------------------------------------------------------------------------
extern "C" void kernel_launch(void* const* d_in, const int* in_sizes, int n_in,
                              void* d_out, int out_size, void* d_ws, size_t ws_size,
                              hipStream_t stream)
{
  (void)in_sizes; (void)n_in; (void)out_size; (void)ws_size;
  const float* wn    = (const float*)d_in[0];
  const int*   signs = (const int*)d_in[1];
  const float* x0    = (const float*)d_in[2];
  const float* y0    = (const float*)d_in[3];
  const float* Wi    = (const float*)d_in[4];
  const float* bi    = (const float*)d_in[5];
  const float* Wh1   = (const float*)d_in[6];
  const float* bh1   = (const float*)d_in[7];
  const float* Wh2   = (const float*)d_in[8];
  const float* bh2   = (const float*)d_in[9];
  const float* Wo    = (const float*)d_in[10];
  const float* bo    = (const float*)d_in[11];

  float* xout = (float*)d_out;                       // [2048*100]
  float* yout = (float*)d_out + BATCH * XDIM;        // [2048]

  char* ws = (char*)d_ws;
  const size_t U_BYTES = (size_t)NSTEP * BATCH * 2 * K0 * 2;   // 128 MB bf16 rows
  __bf16* U    = (__bf16*)ws;
  __bf16* WPK  = (__bf16*)(ws + U_BYTES);                      // 192512 bf16
  float*  part = (float*)(ws + U_BYTES + 192512 * 2);          // 256*2048 f32

  prep_x   <<<(BATCH * DDIM + 255) / 256, 256, 0, stream>>>(wn, signs, x0, U, xout);
  prep_cols<<<(NSTEP * BATCH * 2 + 255) / 256, 256, 0, stream>>>(U);
  prep_w   <<<(192512 + 255) / 256, 256, 0, stream>>>(Wi, Wh1, Wh2, Wo, WPK);
  bsde_main<<<NSTEP * 2 * 16, 256, 0, stream>>>(U, WPK, wn, signs, bi, bh1, bh2, bo, part);
  y_reduce <<<(BATCH + 255) / 256, 256, 0, stream>>>(part, y0, yout);
}
